// SSM_55808805044269
// MI455X (gfx1250) — matrix operand order repaired and checked by reading, not compiled, not run on hardware
//
#include <hip/hip_runtime.h>
#include <stdint.h>

#define L_SEQ 16384
#define CH    512
#define S_DIM 128
#define NT    256   // concatenated [WB | WC] output columns
#define KT_STEPS (CH / 32)

typedef __attribute__((ext_vector_type(16))) __bf16 v16bf;
typedef __attribute__((ext_vector_type(8)))  __bf16 v8bf;
typedef __attribute__((ext_vector_type(8)))  float  v8f;
typedef __attribute__((ext_vector_type(4)))  float  v4f;
typedef __attribute__((ext_vector_type(4)))  unsigned int v4u;
typedef __attribute__((ext_vector_type(8)))  int v8i;
typedef __attribute__((ext_vector_type(4)))  int v4i;

union Frag16 { v16bf v; v8bf h[2]; };

// Compiler fence: tensor_load_to_lds' LDS writes are invisible to LLVM's alias
// analysis; this forces subsequent LDS loads to be materialized (not folded).
#define LDS_PRODUCE_FENCE() asm volatile("" ::: "memory")

// ---- Tensor Data Mover: 2-D tile Global -> LDS (D# packed per ISA 08_async_tensor) ----
// Group0: [1:0]=count=1, [63:32]=lds_addr, [120:64]=global_addr, [127:126]=type=2
// Group1: [17:16]=data_size, [79:48]=tensor_dim0, [111:80]=tensor_dim1,
//         [127:112]=tile_dim0, [143:128]=tile_dim1, [207:160]=tensor_dim0_stride
// This toolchain uses the 6-arg builtin: (v4u, v8i, v4i, v4i, v8i, i32 cpol).
__device__ __forceinline__ void tdm_load_2d(unsigned lds_off, const void* gptr,
                                            unsigned data_size_log2,
                                            unsigned tile_w, unsigned tile_h,
                                            unsigned tensor_w, unsigned tensor_h,
                                            unsigned stride_elems) {
    unsigned long long ga = (unsigned long long)(uintptr_t)gptr;
    v4u g0;
    g0[0] = 1u;                                                 // count=1 (user mode)
    g0[1] = lds_off;                                            // lds_addr (bytes)
    g0[2] = (unsigned)ga;                                       // global_addr[31:0]
    g0[3] = (unsigned)((ga >> 32) & 0x01ffffffu) | (2u << 30);  // addr[56:32] | type=2
    v8i g1;
    g1[0] = (int)(data_size_log2 << 16);                        // wg_mask=0, data_size
    g1[1] = (int)((tensor_w & 0xffffu) << 16);                  // tensor_dim0 lo16
    g1[2] = (int)((tensor_w >> 16) & 0xffffu) | (int)((tensor_h & 0xffffu) << 16);
    g1[3] = (int)((tensor_h >> 16) & 0xffffu) | (int)((tile_w & 0xffffu) << 16);
    g1[4] = (int)(tile_h & 0xffffu);                            // tile_dim1 (dim2=0)
    g1[5] = (int)stride_elems;                                  // dim0_stride lo32
    g1[6] = 0;
    g1[7] = 0;
    v4i z4 = {0, 0, 0, 0};
    v8i z8 = {0, 0, 0, 0, 0, 0, 0, 0};
    __builtin_amdgcn_tensor_load_to_lds(g0, g1, z4, z4, z8, 0);
}

// ---------------- W concat + convert (512x256 bf16) ----------------
__global__ __launch_bounds__(256)
void cvt_w_kernel(const float* __restrict__ WB, const float* __restrict__ WC,
                  __bf16* __restrict__ Wcat) {
    int i = blockIdx.x * 256 + threadIdx.x;
    int k = i >> 8;
    int n = i & 255;
    float v = (n < S_DIM) ? WB[k * S_DIM + n] : WC[k * S_DIM + (n - S_DIM)];
    Wcat[i] = (__bf16)v;
}

// ---------------- WMMA GEMM: BC = x @ [WB|WC], TDM-staged through LDS ----------------
// 512 blocks x 256 threads (8 waves). Block: 32 output rows, full N=256.
// Wave w: m-sub = w&1 (16 rows), n-group = w>>1 (64 cols = 4 WMMA tiles).
// Wave 0 drives TDM: per K-step, A tile = 32x32 f32 of x (converted to bf16 at
// fragment build), B tile = 32x256 bf16 of Wcat. Double buffered: 40KB LDS.
__global__ __launch_bounds__(256)
void gemm_bc_kernel(const float* __restrict__ x, const __bf16* __restrict__ Wcat,
                    float* __restrict__ BC) {
    __shared__ __align__(16) unsigned char shraw[8192 + 32768];
    float*  shA = (float*)shraw;                  // 2 x (32x32 f32)
    __bf16* shB = (__bf16*)(shraw + 8192);        // 2 x (32x256 bf16)

    // LDS byte offsets of the staging buffers (low 32 bits of generic pointer
    // to an LDS object are the offset within the workgroup's LDS allocation).
    const unsigned ldsBase = (unsigned)(uintptr_t)(void*)shraw;
    const unsigned ldsA[2] = { ldsBase,        ldsBase + 4096u  };
    const unsigned ldsB[2] = { ldsBase + 8192u, ldsBase + 24576u };

    const int lane   = threadIdx.x & 31;
    const int wave   = threadIdx.x >> 5;
    const int laneHi = lane >> 4;
    const int laneLo = lane & 15;
    const int mbase  = blockIdx.x * 32;
    const int n0     = (wave >> 1) * 64;
    const int mr     = (wave & 1) * 16 + laneLo;  // local A-tile row

    // wave id is uniform per wave; force a scalar branch
    const bool tdmWave = (__builtin_amdgcn_readfirstlane(wave) == 0);

    v8f acc0 = {}, acc1 = {}, acc2 = {}, acc3 = {};

    if (tdmWave) {
        tdm_load_2d(ldsA[0], x + (size_t)mbase * CH, 2, 32, 32, CH, L_SEQ, CH);
        tdm_load_2d(ldsB[0], Wcat,                   1, NT, 32, NT, CH, NT);
    }

    for (int kt = 0; kt < KT_STEPS; ++kt) {
        const int cur = kt & 1;
        if (tdmWave) __builtin_amdgcn_s_wait_tensorcnt(0);  // stage kt landed
        __syncthreads();                                    // visible to all waves
        LDS_PRODUCE_FENCE();                                // TDM wrote shraw: reload
        if (tdmWave && kt + 1 < KT_STEPS) {                 // prefetch stage kt+1
            const int k0 = (kt + 1) * 32;
            tdm_load_2d(ldsA[cur ^ 1], x + (size_t)mbase * CH + k0, 2, 32, 32, CH, L_SEQ, CH);
            tdm_load_2d(ldsB[cur ^ 1], Wcat + (size_t)k0 * NT,      1, NT, 32, NT, CH, NT);
        }

        // A fragment: row mr, K cols laneHi*8+{0..7} and 16+laneHi*8+{0..7}; f32 -> bf16
        const float* af = shA + cur * 1024 + mr * 32 + laneHi * 8;
        v4f f0 = *(const v4f*)(af);
        v4f f1 = *(const v4f*)(af + 4);
        v4f f2 = *(const v4f*)(af + 16);
        v4f f3 = *(const v4f*)(af + 20);
        Frag16 a;
        #pragma unroll
        for (int i = 0; i < 4; ++i) {
            a.v[i]      = (__bf16)f0[i];
            a.v[4 + i]  = (__bf16)f1[i];
            a.v[8 + i]  = (__bf16)f2[i];
            a.v[12 + i] = (__bf16)f3[i];
        }

        // B fragments: lane = N col laneLo, K rows laneHi*8+{0..7} and 16+laneHi*8+{0..7}
        const __bf16* bb = shB + cur * 8192 + n0 + laneLo;
        Frag16 b0, b1, b2, b3;
        #pragma unroll
        for (int i = 0; i < 8; ++i) {
            const int r0 = (laneHi * 8 + i) * NT;
            const int r1 = (16 + laneHi * 8 + i) * NT;
            b0.v[i] = bb[r0 +  0]; b0.v[8 + i] = bb[r1 +  0];
            b1.v[i] = bb[r0 + 16]; b1.v[8 + i] = bb[r1 + 16];
            b2.v[i] = bb[r0 + 32]; b2.v[8 + i] = bb[r1 + 32];
            b3.v[i] = bb[r0 + 48]; b3.v[8 + i] = bb[r1 + 48];
        }

        acc0 = __builtin_amdgcn_wmma_f32_16x16x32_bf16(false, a.v, false, b0.v, (short)0, acc0, false, false);
        acc1 = __builtin_amdgcn_wmma_f32_16x16x32_bf16(false, a.v, false, b1.v, (short)0, acc1, false, false);
        acc2 = __builtin_amdgcn_wmma_f32_16x16x32_bf16(false, a.v, false, b2.v, (short)0, acc2, false, false);
        acc3 = __builtin_amdgcn_wmma_f32_16x16x32_bf16(false, a.v, false, b3.v, (short)0, acc3, false, false);
    }

    // D layout: VGPR r -> M = r + 8*laneHi, N = laneLo
    float* o = BC + (size_t)(mbase + (wave & 1) * 16 + 8 * laneHi) * NT + n0 + laneLo;
    #pragma unroll
    for (int r = 0; r < 8; ++r) {
        o[(size_t)r * NT +  0] = acc0[r];
        o[(size_t)r * NT + 16] = acc1[r];
        o[(size_t)r * NT + 32] = acc2[r];
        o[(size_t)r * NT + 48] = acc3[r];
    }
}

// ---------------- dt = softplus(x @ Wdt + bdt), one wave per row ----------------
__global__ __launch_bounds__(256)
void dt_kernel(const float* __restrict__ x, const float* __restrict__ Wdt,
               const float* __restrict__ bdt, float* __restrict__ dt) {
    const int lane = threadIdx.x & 31;
    const int wave = threadIdx.x >> 5;
    const int l    = blockIdx.x * 8 + wave;
    const float* xr = x + (size_t)l * CH;
    float s = 0.f;
    #pragma unroll
    for (int j = 0; j < CH / 32; ++j) s += xr[lane + 32 * j] * Wdt[lane + 32 * j];
    #pragma unroll
    for (int o = 16; o > 0; o >>= 1) s += __shfl_xor(s, o, 32);
    if (lane == 0) {
        float t = s + bdt[0];
        dt[l] = (t > 20.f) ? t : log1pf(__expf(t));
    }
}

// ---------------- exclusive prefix sum of dt over L (single block) ----------------
__global__ __launch_bounds__(256)
void scan_kernel(const float* __restrict__ dt, float* __restrict__ dtex) {
    __shared__ float buf[256];
    __shared__ float carry;
    const int tid = threadIdx.x;
    if (tid == 0) carry = 0.f;
    __syncthreads();
    for (int c = 0; c < L_SEQ / 256; ++c) {
        float v = dt[c * 256 + tid];
        buf[tid] = v;
        __syncthreads();
        for (int o = 1; o < 256; o <<= 1) {
            float t = (tid >= o) ? buf[tid - o] : 0.f;
            __syncthreads();
            buf[tid] += t;
            __syncthreads();
        }
        float incl = buf[tid];
        float c0   = carry;
        dtex[c * 256 + tid] = c0 + incl - v;
        __syncthreads();
        if (tid == 255) carry = c0 + incl;
        __syncthreads();
    }
}

// ---------------- out[l,c] = x[l,c] * sum_s C[l,s]*exp(A_s*Dtex_l)*Bbar[l,s] ----------------
__global__ __launch_bounds__(256)
void final_kernel(const float* __restrict__ x, const float* __restrict__ logA,
                  const float* __restrict__ BC, const float* __restrict__ dt,
                  const float* __restrict__ dtex, float* __restrict__ out) {
    const int lane = threadIdx.x & 31;
    const int wave = threadIdx.x >> 5;
    const int l    = blockIdx.x * 8 + wave;
    const float dtl = dt[l];
    const float dl  = dtex[l];
    const float* bc = BC + (size_t)l * NT;
    float acc = 0.f;
    #pragma unroll
    for (int j = 0; j < S_DIM / 32; ++j) {
        int s    = lane + 32 * j;
        float As = -__expf(logA[s]);
        float ab = __expf(As * dtl);
        float bb = (ab - 1.f) * bc[s] / As;
        float p  = __expf(As * dl);
        acc += bc[S_DIM + s] * p * bb;
    }
    #pragma unroll
    for (int o = 16; o > 0; o >>= 1) acc += __shfl_xor(acc, o, 32);
    const float* xr  = x   + (size_t)l * CH;
    float*       orw = out + (size_t)l * CH;
    #pragma unroll
    for (int j = 0; j < CH / 32; ++j) orw[lane + 32 * j] = xr[lane + 32 * j] * acc;
}

// ---------------- launcher ----------------
extern "C" void kernel_launch(void* const* d_in, const int* in_sizes, int n_in,
                              void* d_out, int out_size, void* d_ws, size_t ws_size,
                              hipStream_t stream) {
    const float* x    = (const float*)d_in[0];
    const float* logA = (const float*)d_in[1];
    const float* WB   = (const float*)d_in[2];
    const float* WC   = (const float*)d_in[3];
    const float* Wdt  = (const float*)d_in[4];
    const float* bdt  = (const float*)d_in[5];
    float* out = (float*)d_out;

    char* ws = (char*)d_ws;
    size_t off = 0;
    __bf16* Wcat = (__bf16*)(ws + off); off += (size_t)CH * NT * sizeof(__bf16);
    float*  BC   = (float*) (ws + off); off += (size_t)L_SEQ * NT * sizeof(float);
    float*  dt   = (float*) (ws + off); off += (size_t)L_SEQ * sizeof(float);
    float*  dtex = (float*) (ws + off); off += (size_t)L_SEQ * sizeof(float);
    (void)ws_size; (void)in_sizes; (void)n_in; (void)out_size;

    cvt_w_kernel<<<(CH * NT) / 256, 256, 0, stream>>>(WB, WC, Wcat);
    gemm_bc_kernel<<<L_SEQ / 32, 256, 0, stream>>>(x, Wcat, BC);
    dt_kernel<<<L_SEQ / 8, 256, 0, stream>>>(x, Wdt, bdt, dt);
    scan_kernel<<<1, 256, 0, stream>>>(dt, dtex);
    final_kernel<<<L_SEQ / 8, 256, 0, stream>>>(x, logA, BC, dt, dtex, out);
}
